// ScaledDotProductAttention_18442589569166
// MI455X (gfx1250) — matrix operand order repaired and checked by reading, not compiled, not run on hardware
//
#include <hip/hip_runtime.h>
#include <hip/hip_bf16.h>
#include <cmath>

typedef __attribute__((ext_vector_type(16))) _Float16 v16h;
typedef __attribute__((ext_vector_type(8)))  float    v8f;

#define B_   2
#define H_   16
#define S_   2048
#define D_   64
#define ROWSTRIDE 2052     // 2048 + 4 floats pad -> breaks lo/hi lane-half bank conflicts
#define LUTN  4095         // distances -2047 .. +2047
#define LUTPAD 4096

// T5 bidirectional bucket for n = i - j = -(relative_position):
// num_buckets halved to 16, max_exact=8, max_distance=128
// large: 8 + int(log(n/8)/log(16) * 8) = 8 + int(2*log2(n/8)), capped at 15
__device__ __forceinline__ int rel_bucket_n(int n) {
  int ret = (n < 0) ? 16 : 0;
  n = (n < 0) ? -n : n;
  if (n < 8) return ret + n;
  int vl = 8 + (int)(log2f((float)n * 0.125f) * 2.0f);
  vl = vl < 15 ? vl : 15;
  return ret + vl;
}

extern "C" __global__ __launch_bounds__(128)
void t5_attn_kernel(const float* __restrict__ q,
                    const float* __restrict__ k,
                    const float* __restrict__ v,
                    const float* __restrict__ bw,
                    const int* __restrict__ cpos,
                    const int* __restrict__ mpos,
                    const unsigned char* __restrict__ mask,
                    float* __restrict__ out_o,
                    float* __restrict__ out_attn,
                    float* __restrict__ out_pb)
{
  extern __shared__ float smem[];
  float* s_lut   = smem;                     // LUTPAD floats: distance -> bias value
  float* s_score = smem + LUTPAD;            // 16 x ROWSTRIDE scores for this q-tile
  unsigned char* s_mask =                    // 4 waves x 2 bufs x 256 B mask tiles
      (unsigned char*)(smem + LUTPAD + 16 * ROWSTRIDE);

  const int wg   = blockIdx.x;               // 4096 = 2 * 16 * 128
  const int qt   = wg & 127;
  const int h    = (wg >> 7) & 15;
  const int b    = wg >> 11;
  const int tid  = threadIdx.x;
  const int wave = tid >> 5;
  const int lane = tid & 31;
  const int nn   = lane & 15;                // column / row-within-16 index
  const int half = lane >> 4;

  const int qbase = qt * 16;
  const size_t bh = (size_t)b * H_ + h;
  const float* qtile = q + (bh * S_ + qbase) * D_;
  const float* kbh   = k + bh * (size_t)S_ * D_;
  const float* vbh   = v + bh * (size_t)S_ * D_;
  const unsigned char* mb = mask + (size_t)b * S_ * S_;

  // ---- one-time: distance -> bias LUT (hoists all log/branch work out of hot loop)
  for (int d = tid; d < LUTN; d += 128) {
    int n = d - 2047;                        // n = i - j
    s_lut[d] = bw[rel_bucket_n(n) * H_ + h];
  }
  __syncthreads();

  // ---- Q tile -> WMMA A operands (f16). A swizzle: lane holds row nn,
  //      lo lanes K={0..7,16..23}(+32), hi lanes K={8..15,24..31}(+32) ----
  const float* qrow = qtile + nn * D_;
  const int base0 = half * 8;
  v16h qa1, qa2;
  #pragma unroll
  for (int p = 0; p < 8; ++p) {
    qa1[p]     = (_Float16)qrow[base0 + p];
    qa1[p + 8] = (_Float16)qrow[16 + base0 + p];
    qa2[p]     = (_Float16)qrow[32 + base0 + p];
    qa2[p + 8] = (_Float16)qrow[48 + base0 + p];
  }

  // context positions for the 8 C/D rows this lane owns (M = r + 8*half)
  int ip[8];
  #pragma unroll
  for (int r = 0; r < 8; ++r)
    ip[r] = cpos[b * S_ + qbase + r + half * 8];

  // async-stage a 16x16 mask tile (16 rows x 16 B) into this wave's LDS buffer
  auto stage_mask = [&](int kb2, int sel) {
    if (lane < 16) {
      unsigned ldsa = (unsigned)(uintptr_t)(s_mask + (wave * 2 + sel) * 256 + lane * 16);
      const unsigned char* ga = mb + (size_t)(qbase + lane) * S_ + kb2;
      asm volatile("global_load_async_to_lds_b128 %0, %1, off"
                   :: "v"(ldsa), "v"(ga) : "memory");
    }
  };

  // ---- phase 1: scores = (Q·Kt)/8 + bias, masked -> LDS; emit position_bias ----
  int sel = 0;
  stage_mask(wave * 16, sel);                // preload first tile's mask
  for (int kt = wave; kt < S_ / 16; kt += 4) {
    const int kbase = kt * 16;
    const float* krow = kbh + (size_t)(kbase + nn) * D_;
    if (kt + 4 < S_ / 16)  // prefetch this wave's next K tile (global_prefetch_b8)
      __builtin_prefetch(kbh + (size_t)(kbase + 64 + nn) * D_, 0, 0);

    // B = K^T: lane holds column nn (= K row nn), contraction idx per A swizzle
    v16h kb1, kb2;
    #pragma unroll
    for (int p = 0; p < 8; ++p) {
      kb1[p]     = (_Float16)krow[half * 8 + p];
      kb1[p + 8] = (_Float16)krow[16 + half * 8 + p];
      kb2[p]     = (_Float16)krow[32 + half * 8 + p];
      kb2[p + 8] = (_Float16)krow[48 + half * 8 + p];
    }
    v8f c = {};
    c = __builtin_amdgcn_wmma_f32_16x16x32_f16(false, qa1, false, kb1,
                                               (short)0, c, false, false);
    c = __builtin_amdgcn_wmma_f32_16x16x32_f16(false, qa2, false, kb2,
                                               (short)0, c, false, false);

    // double-buffer: kick off next tile's mask, then wait for current one
    if (kt + 4 < S_ / 16) {
      stage_mask((kt + 4) * 16, sel ^ 1);
      asm volatile("s_wait_asynccnt 0x1" ::: "memory");
    } else {
      asm volatile("s_wait_asynccnt 0x0" ::: "memory");
    }
    const unsigned char* mtile = s_mask + (wave * 2 + sel) * 256;

    const int jpos = mpos[b * S_ + kbase + nn];
    #pragma unroll
    for (int r = 0; r < 8; ++r) {
      const int m = r + half * 8;            // q row within tile (C layout)
      const int i = qbase + m;
      const int j = kbase + nn;
      int d = ip[r] - jpos + 2047;
      d = d < 0 ? 0 : (d > LUTN - 1 ? LUTN - 1 : d);
      const float bias = s_lut[d];
      float sv = c[r] * 0.125f + bias;       // TEMPERATURE = 8
      if (mtile[m * 16 + nn]) sv = -1.0e9f;
      out_pb[(bh * S_ + i) * (size_t)S_ + j] = bias;
      s_score[m * ROWSTRIDE + j] = sv;
    }
    sel ^= 1;
  }
  __syncthreads();

  // ---- phase 2: exact row softmax; write attn; keep normalized P in LDS ----
  for (int m = wave * 4; m < wave * 4 + 4; ++m) {
    float* row = s_score + m * ROWSTRIDE;
    float mx = -3.4e38f;
    for (int idx = lane; idx < S_; idx += 32) mx = fmaxf(mx, row[idx]);
    #pragma unroll
    for (int off = 16; off > 0; off >>= 1)
      mx = fmaxf(mx, __shfl_xor(mx, off, 32));
    float sum = 0.f;
    for (int idx = lane; idx < S_; idx += 32) {
      float e = __expf(row[idx] - mx);
      row[idx] = e;
      sum += e;
    }
    #pragma unroll
    for (int off = 16; off > 0; off >>= 1)
      sum += __shfl_xor(sum, off, 32);
    const float inv = 1.0f / sum;
    float* arow = out_attn + (bh * S_ + (qbase + m)) * (size_t)S_;
    for (int idx = lane; idx < S_; idx += 32) {
      float pv = row[idx] * inv;
      row[idx] = pv;
      arow[idx] = pv;                        // coalesced attn write
    }
  }
  __syncthreads();

  // ---- phase 3: O = P·V ; each wave owns 16 of the 64 output columns ----
  const int dbase = wave * 16;
  v8f co = {};
  for (int kb = 0; kb < S_; kb += 32) {
    v16h pa, vb;
    // A = P tile: lane holds row nn, K chunk per A swizzle
    const float* prow = s_score + nn * ROWSTRIDE + kb + half * 8;
    #pragma unroll
    for (int p = 0; p < 8; ++p) {
      pa[p]     = (_Float16)prow[p];
      pa[p + 8] = (_Float16)prow[16 + p];
    }
    // B = V tile: lane holds column (dbase+nn), contraction rows per A swizzle
    const float* vcol = vbh + (size_t)(kb + half * 8) * D_ + dbase + nn;
    #pragma unroll
    for (int p = 0; p < 8; ++p) {
      vb[p]     = (_Float16)vcol[p * D_];
      vb[p + 8] = (_Float16)vcol[(16 + p) * D_];
    }
    co = __builtin_amdgcn_wmma_f32_16x16x32_f16(false, pa, false, vb,
                                                (short)0, co, false, false);
  }
  #pragma unroll
  for (int r = 0; r < 8; ++r) {
    const int m = r + half * 8;
    out_o[(bh * S_ + (qbase + m)) * (size_t)D_ + dbase + nn] = co[r];
  }
}

extern "C" void kernel_launch(void* const* d_in, const int* in_sizes, int n_in,
                              void* d_out, int out_size, void* d_ws, size_t ws_size,
                              hipStream_t stream) {
  const float* q  = (const float*)d_in[0];
  const float* k  = (const float*)d_in[1];
  const float* v  = (const float*)d_in[2];
  const float* bw = (const float*)d_in[3];
  const int* cp   = (const int*)d_in[4];
  const int* mp   = (const int*)d_in[5];
  const unsigned char* mask = (const unsigned char*)d_in[6];  // bool = 1 byte

  float* out  = (float*)d_out;
  float* attn = out  + (size_t)B_ * H_ * S_ * D_;   // outputs concatenated in return order
  float* pb   = attn + (size_t)B_ * H_ * S_ * S_;

  const size_t shmem = (LUTPAD + 16 * ROWSTRIDE) * sizeof(float)  // LUT + scores
                     + 4 * 2 * 256;                               // mask double-buffers
  dim3 grid(B_ * H_ * (S_ / 16));   // 4096 workgroups
  dim3 block(128);                  // 4 wave32s
  hipLaunchKernelGGL(t5_attn_kernel, grid, block, shmem, stream,
                     q, k, v, bw, cp, mp, mask, out, attn, pb);
}
